// TreeLSTM_layer_util_36215164240832
// MI455X (gfx1250) — matrix operand order repaired and checked by reading, not compiled, not run on hardware
//
#include <hip/hip_runtime.h>

typedef __attribute__((ext_vector_type(16))) __bf16 v16bf;
typedef __attribute__((ext_vector_type(8)))  float  v8f;

#define D_DIM 128
#define K_TOT 384                 // concatenated K: [x[src] | x[dst] | e]
#define LDS_STRIDE 392            // bf16 elems per LDS row: 384 + 8 pad (bank spread)
#define PERM_HI2 0x07060302u      // v_perm selector: pack {a.hi16, b.hi16}

__device__ __forceinline__ unsigned short bf16_rne(float f) {
  unsigned u = __builtin_bit_cast(unsigned, f);
  unsigned r = u + 0x7FFFu + ((u >> 16) & 1u);
  return (unsigned short)(r >> 16);
}
__device__ __forceinline__ float bf16_to_f(unsigned short h) {
  return __builtin_bit_cast(float, ((unsigned)h) << 16);
}

union FragA { unsigned d[8]; uint4 q[2]; v16bf v; };
union FragB { uint4 q[2]; v16bf v; };
union AVec  { float4 f4[4]; float f[16]; };

// ---------------------------------------------------------------------------
// Prep 1: W[384,128] fp32 -> transposed bf16 hi/lo Wt[n][k] (k-contiguous).
// ---------------------------------------------------------------------------
__global__ void prep_w_kernel(const float* __restrict__ W,
                              unsigned short* __restrict__ whi,
                              unsigned short* __restrict__ wlo) {
  int idx = blockIdx.x * blockDim.x + threadIdx.x;
  if (idx >= K_TOT * D_DIM) return;
  int k = idx / D_DIM;
  int n = idx - k * D_DIM;
  float w = W[idx];
  unsigned short h = bf16_rne(w);
  unsigned short l = bf16_rne(w - bf16_to_f(h));
  whi[n * K_TOT + k] = h;
  wlo[n * K_TOT + k] = l;
}

// ---------------------------------------------------------------------------
// Prep 2: x[N,128] fp32 -> bf16 hi/lo (trunc split), packed 2-per-dword.
// ---------------------------------------------------------------------------
__global__ void prep_x_kernel(const float* __restrict__ x,
                              unsigned* __restrict__ xhi,
                              unsigned* __restrict__ xlo, int total2) {
  int i = blockIdx.x * blockDim.x + threadIdx.x;
  if (i >= total2) return;
  float2 f = ((const float2*)x)[i];
  unsigned u0 = __builtin_bit_cast(unsigned, f.x);
  unsigned u1 = __builtin_bit_cast(unsigned, f.y);
  xhi[i] = __builtin_amdgcn_perm(u1, u0, PERM_HI2);
  float l0 = f.x - __builtin_bit_cast(float, u0 & 0xFFFF0000u);
  float l1 = f.y - __builtin_bit_cast(float, u1 & 0xFFFF0000u);
  xlo[i] = __builtin_amdgcn_perm(__builtin_bit_cast(unsigned, l1),
                                 __builtin_bit_cast(unsigned, l0), PERM_HI2);
}

// Cheap in-register trunc split of 16 fp32 -> bf16 hi/lo fragments.
__device__ __forceinline__ void split16(const AVec& av, FragA& ah, FragA& al) {
#pragma unroll
  for (int p = 0; p < 8; ++p) {
    float f0 = av.f[2 * p], f1 = av.f[2 * p + 1];
    unsigned u0 = __builtin_bit_cast(unsigned, f0);
    unsigned u1 = __builtin_bit_cast(unsigned, f1);
    ah.d[p] = __builtin_amdgcn_perm(u1, u0, PERM_HI2);
    float l0 = f0 - __builtin_bit_cast(float, u0 & 0xFFFF0000u);
    float l1 = f1 - __builtin_bit_cast(float, u1 & 0xFFFF0000u);
    al.d[p] = __builtin_amdgcn_perm(__builtin_bit_cast(unsigned, l1),
                                    __builtin_bit_cast(unsigned, l0), PERM_HI2);
  }
}

// Load one A fragment pair (hi/lo) for a given edge at K-step ks.
template <bool PRECONV>
__device__ __forceinline__ void loadA(const float* __restrict__ x,
                                      const unsigned short* __restrict__ xhi,
                                      const unsigned short* __restrict__ xlo,
                                      size_t rS, size_t rD,
                                      const float* __restrict__ rowE,
                                      int ks, int off, FragA& ah, FragA& al) {
  if (PRECONV && ks < 8) {
    const size_t rb = (ks < 4 ? rS : rD);
    const unsigned short* ph = xhi + rb + off;
    const unsigned short* pl = xlo + rb + off;
    ah.q[0] = *(const uint4*)ph;
    ah.q[1] = *(const uint4*)(ph + 16);
    al.q[0] = *(const uint4*)pl;
    al.q[1] = *(const uint4*)(pl + 16);
  } else {
    const float* row = PRECONV ? rowE
                               : (ks < 4 ? x + rS : (ks < 8 ? x + rD : rowE));
    AVec av;
    av.f4[0] = *(const float4*)(row + off);
    av.f4[1] = *(const float4*)(row + off + 4);
    av.f4[2] = *(const float4*)(row + off + 16);
    av.f4[3] = *(const float4*)(row + off + 20);
    split16(av, ah, al);
  }
}

// ---------------------------------------------------------------------------
// Fused gather-GEMM: out[E,128] = [x[src]|x[dst]|e] @ Wc[384,128]
// 512 threads = 16 wave32 waves; each wave owns TWO 16-edge x 128-col strips
// so every B fragment fetched from LDS feeds 8 WMMAs (LDS bw : XDL balanced).
// 4-WMMA bf16 hi/lo split (hh+lh+hl+ll) ~ fp32-class accuracy, memory-bound.
// ---------------------------------------------------------------------------
template <bool PRECONV>
__global__ void __launch_bounds__(512)
gnn_fused_gemm(const float* __restrict__ x, const float* __restrict__ e,
               const unsigned short* __restrict__ xhi,
               const unsigned short* __restrict__ xlo,
               const unsigned short* __restrict__ whi_g,
               const unsigned short* __restrict__ wlo_g,
               const int* __restrict__ src, const int* __restrict__ dst,
               float* __restrict__ out, int E) {
  extern __shared__ unsigned short smem[];            // [2][128][LDS_STRIDE]
  unsigned short* whi = smem;
  unsigned short* wlo = smem + D_DIM * LDS_STRIDE;

  for (int i = threadIdx.x; i < D_DIM * (K_TOT / 8); i += blockDim.x) {
    int n = i / (K_TOT / 8);
    int k = (i - n * (K_TOT / 8)) * 8;
    *(uint4*)(whi + n * LDS_STRIDE + k) = *(const uint4*)(whi_g + n * K_TOT + k);
    *(uint4*)(wlo + n * LDS_STRIDE + k) = *(const uint4*)(wlo_g + n * K_TOT + k);
  }
  __syncthreads();

  const int wave = threadIdx.x >> 5;      // wave32
  const int lane = threadIdx.x & 31;
  const int hl   = lane & 15;
  const int hb   = lane >> 4;

  const int tile0 = (blockIdx.x * (int)(blockDim.x >> 5) + wave) * 32;
  if (tile0 >= E) return;                 // wave-uniform: EXEC stays all-ones
  const bool has2 = (tile0 + 16) < E;     // wave-uniform

  const int edge0 = tile0 + hl;           // E % 16 == 0 -> full tiles
  const int edge1 = min(tile0 + 16 + hl, E - 1);  // clamp keeps gathers legal
  const size_t rS0 = (size_t)src[edge0] * D_DIM;
  const size_t rD0 = (size_t)dst[edge0] * D_DIM;
  const size_t rS1 = (size_t)src[edge1] * D_DIM;
  const size_t rD1 = (size_t)dst[edge1] * D_DIM;
  const float* rowE0 = e + (size_t)edge0 * D_DIM;
  const float* rowE1 = e + (size_t)edge1 * D_DIM;

  // e-segment is consumed last in the K loop: give it lead time.
  __builtin_prefetch(rowE0, 0, 0);
  __builtin_prefetch(rowE0 + 64, 0, 0);
  __builtin_prefetch(rowE1, 0, 0);
  __builtin_prefetch(rowE1 + 64, 0, 0);

  v8f acc0[8] = {};
  v8f acc1[8] = {};

#pragma unroll
  for (int ks = 0; ks < 12; ++ks) {       // K = 384 in steps of 32
    const int off = (ks & 3) * 32 + hb * 8;
    FragA ah0, al0, ah1, al1;
    loadA<PRECONV>(x, xhi, xlo, rS0, rD0, rowE0, ks, off, ah0, al0);
    loadA<PRECONV>(x, xhi, xlo, rS1, rD1, rowE1, ks, off, ah1, al1);

    const int kb = ks * 32 + hb * 8;      // B: lane-half holds K [8h,8h+8) and [16+8h,24+8h)
#pragma unroll
    for (int nt = 0; nt < 8; ++nt) {
      const unsigned short* ph = whi + (nt * 16 + hl) * LDS_STRIDE + kb;
      const unsigned short* pl = wlo + (nt * 16 + hl) * LDS_STRIDE + kb;
      FragB bh, bl;
      bh.q[0] = *(const uint4*)ph;
      bh.q[1] = *(const uint4*)(ph + 16);
      bl.q[0] = *(const uint4*)pl;
      bl.q[1] = *(const uint4*)(pl + 16);
      acc0[nt] = __builtin_amdgcn_wmma_f32_16x16x32_bf16(
          false, ah0.v, false, bh.v, (short)0, acc0[nt], false, false);
      acc0[nt] = __builtin_amdgcn_wmma_f32_16x16x32_bf16(
          false, al0.v, false, bh.v, (short)0, acc0[nt], false, false);
      acc0[nt] = __builtin_amdgcn_wmma_f32_16x16x32_bf16(
          false, ah0.v, false, bl.v, (short)0, acc0[nt], false, false);
      acc0[nt] = __builtin_amdgcn_wmma_f32_16x16x32_bf16(
          false, al0.v, false, bl.v, (short)0, acc0[nt], false, false);
      acc1[nt] = __builtin_amdgcn_wmma_f32_16x16x32_bf16(
          false, ah1.v, false, bh.v, (short)0, acc1[nt], false, false);
      acc1[nt] = __builtin_amdgcn_wmma_f32_16x16x32_bf16(
          false, al1.v, false, bh.v, (short)0, acc1[nt], false, false);
      acc1[nt] = __builtin_amdgcn_wmma_f32_16x16x32_bf16(
          false, ah1.v, false, bl.v, (short)0, acc1[nt], false, false);
      acc1[nt] = __builtin_amdgcn_wmma_f32_16x16x32_bf16(
          false, al1.v, false, bl.v, (short)0, acc1[nt], false, false);
    }
  }

  // C/D layout: VGPR r, lanes 0-15 -> row r, lanes 16-31 -> row r+8; col = hl.
  float* ob0 = out + (size_t)tile0 * D_DIM;
#pragma unroll
  for (int nt = 0; nt < 8; ++nt) {
#pragma unroll
    for (int r = 0; r < 8; ++r) {
      ob0[(size_t)(r + hb * 8) * D_DIM + nt * 16 + hl] = acc0[nt][r];
    }
  }
  if (has2) {                             // wave-uniform guard
    float* ob1 = out + (size_t)(tile0 + 16) * D_DIM;
#pragma unroll
    for (int nt = 0; nt < 8; ++nt) {
#pragma unroll
      for (int r = 0; r < 8; ++r) {
        ob1[(size_t)(r + hb * 8) * D_DIM + nt * 16 + hl] = acc1[nt][r];
      }
    }
  }
}

extern "C" void kernel_launch(void* const* d_in, const int* in_sizes, int n_in,
                              void* d_out, int out_size, void* d_ws, size_t ws_size,
                              hipStream_t stream) {
  (void)n_in; (void)out_size;
  const float* x = (const float*)d_in[0];
  const float* e = (const float*)d_in[1];
  const float* W = (const float*)d_in[2];
  const int*  src = (const int*)d_in[3];
  const int*  dst = (const int*)d_in[4];
  const int E = in_sizes[3];
  const int N = in_sizes[0] / D_DIM;

  unsigned short* whi = (unsigned short*)d_ws;
  unsigned short* wlo = whi + (size_t)K_TOT * D_DIM;
  unsigned short* xhi = wlo + (size_t)K_TOT * D_DIM;
  unsigned short* xlo = xhi + (size_t)N * D_DIM;

  const size_t wBytes = (size_t)2 * K_TOT * D_DIM * sizeof(unsigned short);
  const size_t xBytes = (size_t)2 * N * D_DIM * sizeof(unsigned short);
  const bool preconv = ws_size >= wBytes + xBytes;   // deterministic per run

  {
    int total = K_TOT * D_DIM;
    hipLaunchKernelGGL(prep_w_kernel, dim3((total + 255) / 256), dim3(256), 0,
                       stream, W, whi, wlo);
  }

  const size_t ldsBytes = 2u * D_DIM * LDS_STRIDE * sizeof(unsigned short); // ~196KB
  const int edgesPerBlock = (512 / 32) * 32;          // 512 (dual strip / wave)
  const int blocks = (E + edgesPerBlock - 1) / edgesPerBlock;

  if (preconv) {
    int total2 = N * (D_DIM / 2);
    hipLaunchKernelGGL(prep_x_kernel, dim3((total2 + 255) / 256), dim3(256), 0,
                       stream, x, (unsigned*)xhi, (unsigned*)xlo, total2);
    hipFuncSetAttribute((const void*)gnn_fused_gemm<true>,
                        hipFuncAttributeMaxDynamicSharedMemorySize, (int)ldsBytes);
    hipLaunchKernelGGL(gnn_fused_gemm<true>, dim3(blocks), dim3(512), ldsBytes,
                       stream, x, e, xhi, xlo, whi, wlo, src, dst,
                       (float*)d_out, E);
  } else {
    hipFuncSetAttribute((const void*)gnn_fused_gemm<false>,
                        hipFuncAttributeMaxDynamicSharedMemorySize, (int)ldsBytes);
    hipLaunchKernelGGL(gnn_fused_gemm<false>, dim3(blocks), dim3(512), ldsBytes,
                       stream, x, e, (const unsigned short*)nullptr,
                       (const unsigned short*)nullptr, whi, wlo, src, dst,
                       (float*)d_out, E);
  }
}
